// SemanticIDFramework_45268955299926
// MI455X (gfx1250) — matrix operand order repaired and checked by reading, not compiled, not run on hardware
//
#include <hip/hip_runtime.h>
#include <hip/hip_bf16.h>

// ---------------------------------------------------------------------------
// CDNA5 / gfx1250 WMMA types & helpers
// ---------------------------------------------------------------------------
typedef _Float16 v16h __attribute__((ext_vector_type(16)));
typedef float    v8f  __attribute__((ext_vector_type(8)));
typedef _Float16 f16x2 __attribute__((ext_vector_type(2)));
typedef _Float16 f16x4 __attribute__((ext_vector_type(4)));
typedef int      v2i  __attribute__((ext_vector_type(2)));

template <class A, class B> struct same_t { static constexpr bool v = false; };
template <class A> struct same_t<A, A> { static constexpr bool v = true; };

__device__ __forceinline__ v8f wmma32(v16h a, v16h b, v8f c) {
  return __builtin_amdgcn_wmma_f32_16x16x32_f16(false, a, false, b,
                                                (short)0, c, false, false);
}

__device__ __forceinline__ f16x4 cvt4(float4 v) {
  f16x4 h;
  h[0] = (_Float16)v.x; h[1] = (_Float16)v.y;
  h[2] = (_Float16)v.z; h[3] = (_Float16)v.w;
  return h;
}

// ---- 8-byte global->LDS copy: async on gfx1250 (ASYNCcnt-tracked) ----------
#if __has_builtin(__builtin_amdgcn_global_load_async_to_lds_b64)
#define HAVE_ASYNC_LDS 1
typedef __attribute__((address_space(1))) v2i as1_v2i;
typedef __attribute__((address_space(3))) v2i as3_v2i;
__device__ __forceinline__ void lds_copy4(_Float16* dst, const _Float16* src) {
  __builtin_amdgcn_global_load_async_to_lds_b64(
      (as1_v2i*)(uintptr_t)src,
      (as3_v2i*)(unsigned int)(uintptr_t)dst, 0, 0);
}
template <int N>
__device__ __forceinline__ void lds_wait_n() {
#if __has_builtin(__builtin_amdgcn_s_wait_asynccnt)
  __builtin_amdgcn_s_wait_asynccnt(N);
#else
  asm volatile("s_wait_asynccnt %0" ::"i"(N) : "memory");
#endif
}
#else
#define HAVE_ASYNC_LDS 0
__device__ __forceinline__ void lds_copy4(_Float16* dst, const _Float16* src) {
  *(f16x4*)dst = *(const f16x4*)src;
}
template <int N> __device__ __forceinline__ void lds_wait_n() {}
#endif

// A-fragment (16x32 f16), contiguous k = 16*(lane/16) + s (LDS row-major [row][k]).
__device__ __forceinline__ v16h load_a_frag(const _Float16* base, int ld,
                                            int row0, int k0, int lane) {
  const int m   = lane & 15;
  const int kb8 = (lane & 16) ? 16 : 0;
  const _Float16* p = base + (row0 + m) * ld + k0;
  v16h a;
#pragma unroll
  for (int j = 0; j < 4; ++j) {
    f16x2 lo = *(const f16x2*)(p + kb8 + 2 * j);
    f16x2 hi = *(const f16x2*)(p + 8 + kb8 + 2 * j);
    a[2 * j] = lo[0]; a[2 * j + 1] = lo[1];
    a[8 + 2 * j] = hi[0]; a[8 + 2 * j + 1] = hi[1];
  }
  return a;
}

// B-fragment (32x16 f16), LDS stored [n][k].
__device__ __forceinline__ v16h load_b_frag(const _Float16* base, int ld,
                                            int n0, int k0, int lane) {
  const int n    = lane & 15;
  const int koff = (lane & 16) ? 16 : 0;
  const _Float16* p = base + (n0 + n) * ld + k0 + koff;
  v16h b;
#pragma unroll
  for (int j = 0; j < 8; ++j) {
    f16x2 d2 = *(const f16x2*)(p + 2 * j);
    b[2 * j] = d2[0]; b[2 * j + 1] = d2[1];
  }
  return b;
}

// B-fragment where source LDS is [k][n] (V tiles): scalar gathers.
__device__ __forceinline__ v16h load_b_frag_kn(const _Float16* base, int ld,
                                               int k0, int n0, int lane) {
  const int n    = lane & 15;
  const int koff = (lane & 16) ? 16 : 0;
  v16h b;
#pragma unroll
  for (int j = 0; j < 16; ++j) {
    int kk = k0 + koff + 2 * (j >> 1) + (j & 1);
    b[j] = base[kk * ld + n0 + n];
  }
  return b;
}

// ---------------------------------------------------------------------------
// Generic WMMA GEMM, double-buffered async pipeline:
//   C[M,N] = epi(A[M,K] @ W[N,K]^T + bias [+ resid])
// TA: float (convert at stage) or _Float16 (async copy). W f16. TC f32/f16.
// epi: 0=bias 1=+relu 2=+resid(f16) 3=threshold
// 256 thr = 8 waves; tile 256x64; wave = 64x32 (4x2 frags).
// ---------------------------------------------------------------------------
template <typename TA, typename TC>
__global__ __launch_bounds__(256) void gemm_wmma_kernel(
    const TA* __restrict__ A, const _Float16* __restrict__ W,
    const float* __restrict__ bias, const _Float16* __restrict__ resid,
    TC* __restrict__ C, int M, int N, int Kfull, int ldc, int epi) {
  __shared__ _Float16 As[2][256][40];
  __shared__ _Float16 Bs[2][64][40];
  const int tid = threadIdx.x;
  const int lane = tid & 31, wave = tid >> 5;
  const int wm = wave >> 1, wn = wave & 1;
  const int m0 = blockIdx.y * 256, n0 = blockIdx.x * 64;
  // per-wave async ops per tile (for pipelined s_wait_asynccnt)
  constexpr int INFLIGHT = same_t<TA, float>::v ? 2 : 10;

  const v8f vzero = {0.f, 0.f, 0.f, 0.f, 0.f, 0.f, 0.f, 0.f};
  v8f acc[4][2];
#pragma unroll
  for (int i = 0; i < 4; ++i)
#pragma unroll
    for (int j = 0; j < 2; ++j) acc[i][j] = vzero;

  auto stage = [&](int buf, int kb) {
    if constexpr (same_t<TA, float>::v) {  // f32 A: convert at stage
      const float4* src = (const float4*)(A + (size_t)(m0 + tid) * Kfull + kb);
      if (kb + 32 < Kfull) __builtin_prefetch(src + 8, 0, 1);
#pragma unroll
      for (int j = 0; j < 8; ++j) *(f16x4*)&As[buf][tid][4 * j] = cvt4(src[j]);
    } else {  // f16 A: async copy
      const _Float16* src = A + (size_t)(m0 + tid) * Kfull + kb;
#pragma unroll
      for (int j = 0; j < 8; ++j) lds_copy4(&As[buf][tid][4 * j], src + 4 * j);
    }
    int n = tid >> 2;
    int c0 = (tid & 3) * 8;
    const _Float16* bsrc = W + (size_t)(n0 + n) * Kfull + kb + c0;
    lds_copy4(&Bs[buf][n][c0], bsrc);
    lds_copy4(&Bs[buf][n][c0 + 4], bsrc + 4);
  };

  auto compute = [&](int buf) {
    v16h af[4], bf[2];
#pragma unroll
    for (int i = 0; i < 4; ++i)
      af[i] = load_a_frag(&As[buf][0][0], 40, wm * 64 + i * 16, 0, lane);
#pragma unroll
    for (int j = 0; j < 2; ++j)
      bf[j] = load_b_frag(&Bs[buf][0][0], 40, wn * 32 + j * 16, 0, lane);
#pragma unroll
    for (int i = 0; i < 4; ++i)
#pragma unroll
      for (int j = 0; j < 2; ++j) acc[i][j] = wmma32(af[i], bf[j], acc[i][j]);
  };

  const int nk = Kfull >> 5;
  stage(0, 0);
  for (int it = 0; it < nk; ++it) {
    const int cur = it & 1;
    if (it + 1 < nk) {
      stage(1 - cur, (it + 1) << 5);
      lds_wait_n<INFLIGHT>();  // tile `it` landed; next tile may be in flight
    } else {
      lds_wait_n<0>();
    }
    __syncthreads();
    compute(cur);
    __syncthreads();  // all reads of buf `cur` done before it is restaged
  }

  const int mh = (lane & 16) ? 8 : 0;
  const int nn = lane & 15;
#pragma unroll
  for (int i = 0; i < 4; ++i)
#pragma unroll
    for (int j = 0; j < 2; ++j)
#pragma unroll
      for (int e = 0; e < 8; ++e) {
        int row = m0 + wm * 64 + i * 16 + e + mh;
        int col = n0 + wn * 32 + j * 16 + nn;
        float v = acc[i][j][e];
        if (bias) v += bias[col];
        if (epi == 1) v = fmaxf(v, 0.f);
        else if (epi == 2) v += (float)resid[(size_t)row * ldc + col];
        else if (epi == 3) v = v > 0.f ? 1.f : 0.f;
        C[(size_t)row * ldc + col] = (TC)v;
      }
}

// ---------------------------------------------------------------------------
// Conv1d ('same' pad), tap-major, double-buffered async pipeline:
//   y = relu(sum_t shift(x,t-pad) @ Wt^T + b)
// x f16 [B*S,Din] (S=512), wt f16 tap-major [ktap][O][Din], out f16 (ld ldc).
// ---------------------------------------------------------------------------
__global__ __launch_bounds__(256) void conv_wmma_kernel(
    const _Float16* __restrict__ x, const _Float16* __restrict__ wt,
    const float* __restrict__ bias, _Float16* __restrict__ C,
    int Din, int odim, int ldc, int ktap) {
  __shared__ _Float16 As[2][256][40];
  __shared__ _Float16 Bs[2][64][40];
  const int tid = threadIdx.x;
  const int lane = tid & 31, wave = tid >> 5;
  const int wm = wave >> 1, wn = wave & 1;
  const int m0 = blockIdx.y * 256, n0 = blockIdx.x * 64;
  const int pad = ktap >> 1;

  const v8f vzero = {0.f, 0.f, 0.f, 0.f, 0.f, 0.f, 0.f, 0.f};
  v8f acc[4][2];
#pragma unroll
  for (int i = 0; i < 4; ++i)
#pragma unroll
    for (int j = 0; j < 2; ++j) acc[i][j] = vzero;

  const int gm = m0 + tid;
  const int s = gm & 511;  // S = 512

  auto stage = [&](int buf, int t, int kb) {
    const int delta = t - pad;
    const bool ok = (unsigned)(s + delta) < 512u;
    if (ok) {  // every wave has >=1 in-bounds lane (pad <= 4 < 32)
      const _Float16* src = x + (size_t)(gm + delta) * Din + kb;
#pragma unroll
      for (int j = 0; j < 8; ++j) lds_copy4(&As[buf][tid][4 * j], src + 4 * j);
    } else {
      const f16x4 hz = {(_Float16)0.f, (_Float16)0.f,
                        (_Float16)0.f, (_Float16)0.f};
#pragma unroll
      for (int j = 0; j < 8; ++j) *(f16x4*)&As[buf][tid][4 * j] = hz;
    }
    int n = tid >> 2;
    int c0 = (tid & 3) * 8;
    const _Float16* bsrc =
        wt + (size_t)t * odim + (size_t)(n0 + n) * Din + kb + c0;
    lds_copy4(&Bs[buf][n][c0], bsrc);
    lds_copy4(&Bs[buf][n][c0 + 4], bsrc + 4);
  };

  auto compute = [&](int buf) {
    v16h af[4], bf[2];
#pragma unroll
    for (int i = 0; i < 4; ++i)
      af[i] = load_a_frag(&As[buf][0][0], 40, wm * 64 + i * 16, 0, lane);
#pragma unroll
    for (int j = 0; j < 2; ++j)
      bf[j] = load_b_frag(&Bs[buf][0][0], 40, wn * 32 + j * 16, 0, lane);
#pragma unroll
    for (int i = 0; i < 4; ++i)
#pragma unroll
      for (int j = 0; j < 2; ++j) acc[i][j] = wmma32(af[i], bf[j], acc[i][j]);
  };

  const int niter = ktap * (Din >> 5);
  int t_cur = 0, kb_cur = 0;
  stage(0, 0, 0);
  for (int it = 0; it < niter; ++it) {
    const int cur = it & 1;
    int kb_next = kb_cur + 32, t_next = t_cur;
    if (kb_next == Din) { kb_next = 0; ++t_next; }
    if (it + 1 < niter) {
      stage(1 - cur, t_next, kb_next);
      lds_wait_n<10>();
    } else {
      lds_wait_n<0>();
    }
    __syncthreads();
    compute(cur);
    __syncthreads();
    kb_cur = kb_next;
    t_cur = t_next;
  }

  const int mh = (lane & 16) ? 8 : 0;
  const int nn = lane & 15;
#pragma unroll
  for (int i = 0; i < 4; ++i)
#pragma unroll
    for (int j = 0; j < 2; ++j)
#pragma unroll
      for (int e = 0; e < 8; ++e) {
        int row = m0 + wm * 64 + i * 16 + e + mh;
        int col = n0 + wn * 32 + j * 16 + nn;
        float v = fmaxf(acc[i][j][e] + bias[col], 0.f);
        C[(size_t)row * ldc + col] = (_Float16)v;
      }
}

// ---------------------------------------------------------------------------
// Flash-style MHA. qkv f16 [B*S,1536], out f16 [B*S,512].
// grid = B*H*(S/64), 128 thr (4 waves); wave owns 16 query rows.
// ---------------------------------------------------------------------------
__global__ __launch_bounds__(128) void attn_wmma_kernel(
    const _Float16* __restrict__ qkv, _Float16* __restrict__ out) {
  __shared__ _Float16 Kb[64][72];
  __shared__ _Float16 Vb[64][72];
  __shared__ _Float16 Ps[4][16][72];
  const int tid = threadIdx.x, lane = tid & 31, wave = tid >> 5;
  const int bid = blockIdx.x;
  const int qt = bid & 7, h = (bid >> 3) & 7, b = bid >> 6;
  const int q0 = qt * 64 + wave * 16;
  const int mh = (lane & 16) ? 8 : 0;
  const int nn = lane & 15;

  {  // stage Q (pure copy; softmax scale applied to f32 scores)
    int r = lane >> 1, c0 = (lane & 1) * 32;
    const _Float16* src =
        qkv + ((size_t)(b * 512 + q0 + r)) * 1536 + h * 64 + c0;
#pragma unroll
    for (int j = 0; j < 8; ++j) lds_copy4(&Ps[wave][r][c0 + 4 * j], src + 4 * j);
  }
  lds_wait_n<0>();
  __syncthreads();
  v16h aq0 = load_a_frag(&Ps[wave][0][0], 72, 0, 0, lane);
  v16h aq1 = load_a_frag(&Ps[wave][0][0], 72, 0, 32, lane);

  const v8f vzero = {0.f, 0.f, 0.f, 0.f, 0.f, 0.f, 0.f, 0.f};
  v8f o[4];
#pragma unroll
  for (int nt = 0; nt < 4; ++nt) o[nt] = vzero;
  float mi[8], li[8];
#pragma unroll
  for (int e = 0; e < 8; ++e) { mi[e] = -3.0e38f; li[e] = 0.f; }

  for (int sb = 0; sb < 8; ++sb) {
    __syncthreads();
    {  // stage K/V block (pure copies)
      int r = tid >> 1, c0 = (tid & 1) * 32;
      const _Float16* kk =
          qkv + ((size_t)(b * 512 + sb * 64 + r)) * 1536 + 512 + h * 64 + c0;
      const _Float16* vv = kk + 512;
#pragma unroll
      for (int j = 0; j < 8; ++j) {
        lds_copy4(&Kb[r][c0 + 4 * j], kk + 4 * j);
        lds_copy4(&Vb[r][c0 + 4 * j], vv + 4 * j);
      }
    }
    lds_wait_n<0>();
    __syncthreads();

    // S = (Q @ K^T) * dh^-0.5
    v8f sf[4];
#pragma unroll
    for (int nt = 0; nt < 4; ++nt) {
      v8f a = vzero;
      a = wmma32(aq0, load_b_frag(&Kb[0][0], 72, nt * 16, 0, lane), a);
      a = wmma32(aq1, load_b_frag(&Kb[0][0], 72, nt * 16, 32, lane), a);
#pragma unroll
      for (int e = 0; e < 8; ++e) a[e] *= 0.125f;
      sf[nt] = a;
    }

    // online softmax (rows live in 16-lane groups on wave32)
#pragma unroll
    for (int e = 0; e < 8; ++e) {
      float mx = fmaxf(fmaxf(sf[0][e], sf[1][e]), fmaxf(sf[2][e], sf[3][e]));
#pragma unroll
      for (int m = 1; m < 16; m <<= 1) mx = fmaxf(mx, __shfl_xor(mx, m));
      float mn = fmaxf(mi[e], mx);
      float sc = __expf(mi[e] - mn);
      mi[e] = mn;
      float rs = 0.f;
#pragma unroll
      for (int nt = 0; nt < 4; ++nt) {
        float p = __expf(sf[nt][e] - mn);
        sf[nt][e] = p;
        rs += p;
      }
#pragma unroll
      for (int m = 1; m < 16; m <<= 1) rs += __shfl_xor(rs, m);
      li[e] = li[e] * sc + rs;
#pragma unroll
      for (int nt = 0; nt < 4; ++nt) o[nt][e] *= sc;
    }

    // restage P into A-fragment layout
#pragma unroll
    for (int nt = 0; nt < 4; ++nt)
#pragma unroll
      for (int e = 0; e < 8; ++e)
        Ps[wave][e + mh][nt * 16 + nn] = (_Float16)sf[nt][e];
    __syncthreads();

    v16h ap0 = load_a_frag(&Ps[wave][0][0], 72, 0, 0, lane);
    v16h ap1 = load_a_frag(&Ps[wave][0][0], 72, 0, 32, lane);
#pragma unroll
    for (int nt = 0; nt < 4; ++nt) {
      o[nt] = wmma32(ap0, load_b_frag_kn(&Vb[0][0], 72, 0, nt * 16, lane), o[nt]);
      o[nt] = wmma32(ap1, load_b_frag_kn(&Vb[0][0], 72, 32, nt * 16, lane), o[nt]);
    }
  }

#pragma unroll
  for (int nt = 0; nt < 4; ++nt)
#pragma unroll
    for (int e = 0; e < 8; ++e) {
      size_t row = (size_t)b * 512 + q0 + e + mh;
      out[row * 512 + h * 64 + nt * 16 + nn] = (_Float16)(o[nt][e] / li[e]);
    }
}

// ---------------------------------------------------------------------------
// VQ: argmin_k ( ||c_k||^2 - 2 * enc . c_k ).  enc f32, codebook f16.
// ---------------------------------------------------------------------------
__global__ __launch_bounds__(256) void vq_argmin_kernel(
    const float* __restrict__ enc, const _Float16* __restrict__ cbh,
    const float* __restrict__ cbn, int* __restrict__ idxo,
    int Kcode, int D) {
  __shared__ _Float16 As[128][40];
  __shared__ _Float16 Bs[64][40];
  const int tid = threadIdx.x, lane = tid & 31, wave = tid >> 5;
  const int wm = wave >> 1, wn = wave & 1;
  const int m0 = blockIdx.x * 128;
  const int mh = (lane & 16) ? 8 : 0;
  const int nn = lane & 15;
  const v8f vzero = {0.f, 0.f, 0.f, 0.f, 0.f, 0.f, 0.f, 0.f};

  float best[2][8];
  int bidx[2][8];
#pragma unroll
  for (int i = 0; i < 2; ++i)
#pragma unroll
    for (int e = 0; e < 8; ++e) { best[i][e] = 3.0e38f; bidx[i][e] = 0; }

  for (int ct = 0; ct < Kcode; ct += 64) {
    v8f acc[2][2];
#pragma unroll
    for (int i = 0; i < 2; ++i)
#pragma unroll
      for (int j = 0; j < 2; ++j) acc[i][j] = vzero;

    for (int kb = 0; kb < D; kb += 32) {
      __syncthreads();
      {
        int r = tid >> 1, c0 = (tid & 1) * 16;
        const float4* src =
            (const float4*)(enc + (size_t)(m0 + r) * D + kb + c0);
#pragma unroll
        for (int j = 0; j < 4; ++j) *(f16x4*)&As[r][c0 + 4 * j] = cvt4(src[j]);
      }
      {
        int n = tid >> 2, c0 = (tid & 3) * 8;
        const _Float16* src = cbh + (size_t)(ct + n) * D + kb + c0;
        lds_copy4(&Bs[n][c0], src);
        lds_copy4(&Bs[n][c0 + 4], src + 4);
      }
      lds_wait_n<0>();
      __syncthreads();
      v16h af[2], bf[2];
#pragma unroll
      for (int i = 0; i < 2; ++i)
        af[i] = load_a_frag(&As[0][0], 40, wm * 32 + i * 16, 0, lane);
#pragma unroll
      for (int j = 0; j < 2; ++j)
        bf[j] = load_b_frag(&Bs[0][0], 40, wn * 32 + j * 16, 0, lane);
#pragma unroll
      for (int i = 0; i < 2; ++i)
#pragma unroll
        for (int j = 0; j < 2; ++j) acc[i][j] = wmma32(af[i], bf[j], acc[i][j]);
    }
#pragma unroll
    for (int i = 0; i < 2; ++i)
#pragma unroll
      for (int j = 0; j < 2; ++j)
#pragma unroll
        for (int e = 0; e < 8; ++e) {
          int col = ct + wn * 32 + j * 16 + nn;
          float d = cbn[col] - 2.f * acc[i][j][e];
          if (d < best[i][e]) { best[i][e] = d; bidx[i][e] = col; }
        }
  }

#pragma unroll
  for (int i = 0; i < 2; ++i)
#pragma unroll
    for (int e = 0; e < 8; ++e) {
      float bv = best[i][e];
      int bi = bidx[i][e];
#pragma unroll
      for (int m = 1; m < 16; m <<= 1) {
        float ov = __shfl_xor(bv, m);
        int oi = __shfl_xor(bi, m);
        if (ov < bv || (ov == bv && oi < bi)) { bv = ov; bi = oi; }
      }
      if ((lane & 15) == 0) idxo[m0 + wm * 32 + i * 16 + e + mh] = bi;
    }
}

// ---------------------------------------------------------------------------
// Small supporting kernels
// ---------------------------------------------------------------------------
__global__ void cvt_f16_kernel(const float* __restrict__ in,
                               _Float16* __restrict__ out, int n) {
  int i = blockIdx.x * 256 + threadIdx.x;
  if (i < n) out[i] = (_Float16)in[i];
}

// conv weight [O][Din][ktap] f32 -> tap-major [ktap][O*Din] f16
__global__ void convw_kernel(const float* __restrict__ in,
                             _Float16* __restrict__ out, int odim, int ktap) {
  int i = blockIdx.x * 256 + threadIdx.x;  // o*Din + ic
  if (i < odim) {
    for (int t = 0; t < ktap; ++t)
      out[(size_t)t * odim + i] = (_Float16)in[(size_t)i * ktap + t];
  }
}

__global__ void add_pos_kernel(_Float16* __restrict__ x,
                               const float* __restrict__ pos) {
  size_t i = (size_t)blockIdx.x * blockDim.x + threadIdx.x;
  int d = (int)(i & 511);
  size_t m = i >> 9;
  int s = (int)(m & 511);
  x[i] = (_Float16)((float)x[i] + pos[((size_t)s << 9) | d]);
}

__global__ __launch_bounds__(256) void cbnorm_kernel(
    const float* __restrict__ cb, float* __restrict__ out, int D) {
  __shared__ float red[256];
  int row = blockIdx.x, tid = threadIdx.x;
  const float* p = cb + (size_t)row * D;
  float s = 0.f;
  for (int d = tid; d < D; d += 256) { float v = p[d]; s += v * v; }
  red[tid] = s; __syncthreads();
  for (int k = 128; k > 0; k >>= 1) {
    if (tid < k) red[tid] += red[tid + k];
    __syncthreads();
  }
  if (tid == 0) out[row] = red[0];
}

__global__ void zero_kernel(float* p) { *p = 0.f; }

__global__ __launch_bounds__(256) void vq_gather_kernel(
    const float* __restrict__ enc, const float* __restrict__ cb,
    const int* __restrict__ idx, float* __restrict__ qout,
    float* __restrict__ idxf, float* __restrict__ loss, int D, float lscale) {
  __shared__ float red[256];
  int row = blockIdx.x, tid = threadIdx.x;
  int id = idx[row];
  const float* c = cb + (size_t)id * D;
  const float* e = enc + (size_t)row * D;
  float* q = qout + (size_t)row * D;
  float s = 0.f;
  for (int d = tid; d < D; d += 256) {
    float v = c[d];
    q[d] = v;
    float df = v - e[d];
    s += df * df;
  }
  red[tid] = s; __syncthreads();
  for (int k = 128; k > 0; k >>= 1) {
    if (tid < k) red[tid] += red[tid + k];
    __syncthreads();
  }
  if (tid == 0) {
    atomicAdd(loss, red[0] * lscale);
    idxf[row] = (float)id;
  }
}

__global__ __launch_bounds__(256) void softmax_row_kernel(float* __restrict__ x,
                                                          int N) {
  __shared__ float red[256];
  int tid = threadIdx.x;
  float* p = x + (size_t)blockIdx.x * N;
  float mx = -3.0e38f;
  for (int j = tid; j < N; j += 256) mx = fmaxf(mx, p[j]);
  red[tid] = mx; __syncthreads();
  for (int k = 128; k > 0; k >>= 1) {
    if (tid < k) red[tid] = fmaxf(red[tid], red[tid + k]);
    __syncthreads();
  }
  mx = red[0]; __syncthreads();
  float s = 0.f;
  for (int j = tid; j < N; j += 256) {
    float e = __expf(p[j] - mx);
    p[j] = e;
    s += e;
  }
  red[tid] = s; __syncthreads();
  for (int k = 128; k > 0; k >>= 1) {
    if (tid < k) red[tid] += red[tid + k];
    __syncthreads();
  }
  float inv = 1.f / red[0];
  for (int j = tid; j < N; j += 256) p[j] *= inv;
}

// ---------------------------------------------------------------------------
// Launch: full pipeline.  B=32 S=512 D=512 K=2048 H=8 dh=64
// ---------------------------------------------------------------------------
extern "C" void kernel_launch(void* const* d_in, const int* in_sizes, int n_in,
                              void* d_out, int out_size, void* d_ws,
                              size_t ws_size, hipStream_t stream) {
  (void)in_sizes; (void)n_in; (void)out_size; (void)ws_size;
  const float* market = (const float*)d_in[0];
  const float* W_in   = (const float*)d_in[1];
  const float* b_in   = (const float*)d_in[2];
  const float* pos    = (const float*)d_in[3];
  const float* c3w = (const float*)d_in[4];  const float* c3b = (const float*)d_in[5];
  const float* c5w = (const float*)d_in[6];  const float* c5b = (const float*)d_in[7];
  const float* c7w = (const float*)d_in[8];  const float* c7b = (const float*)d_in[9];
  const float* c9w = (const float*)d_in[10]; const float* c9b = (const float*)d_in[11];
  const float* msw = (const float*)d_in[12]; const float* msb = (const float*)d_in[13];
  const float* ipw = (const float*)d_in[14]; const float* ipb = (const float*)d_in[15];
  const float* aow = (const float*)d_in[16]; const float* aob = (const float*)d_in[17];
  const float* codebook = (const float*)d_in[18];
  const float* hw  = (const float*)d_in[19];
  const float* r1w = (const float*)d_in[20]; const float* r1b = (const float*)d_in[21];
  const float* r2w = (const float*)d_in[22]; const float* r2b = (const float*)d_in[23];

  const int M = 32 * 512;  // 16384 tokens

  // ---- workspace layout (bytes) ----
  char* wsb = (char*)d_ws;
  _Float16* xh    = (_Float16*)(wsb);              // 8.4M f16 (x / mha / h)
  _Float16* multh = (_Float16*)(wsb + 16777216);   // 33.5M f16 (multi / qkv)
  float*    encf  = (float*)(wsb + 16777216);      // enc f32 overlays multh
  _Float16* projh = (_Float16*)(wsb + 83886080);   // 8.4M f16
  char* wp = wsb + 100663296;                      // f16 weight arena
  _Float16* winh  = (_Float16*)wp; wp += (size_t)262144 * 2;
  _Float16* c3t   = (_Float16*)wp; wp += (size_t)786432 * 2;
  _Float16* c5t   = (_Float16*)wp; wp += (size_t)1310720 * 2;
  _Float16* c7t   = (_Float16*)wp; wp += (size_t)1835008 * 2;
  _Float16* c9t   = (_Float16*)wp; wp += (size_t)2359296 * 2;
  _Float16* mswh  = (_Float16*)wp; wp += (size_t)1048576 * 2;
  _Float16* ipwh  = (_Float16*)wp; wp += (size_t)786432 * 2;
  _Float16* aowh  = (_Float16*)wp; wp += (size_t)262144 * 2;
  _Float16* cbh   = (_Float16*)wp; wp += (size_t)1048576 * 2;
  _Float16* hwh   = (_Float16*)wp; wp += (size_t)32768 * 2;
  _Float16* r1h   = (_Float16*)wp; wp += (size_t)131072 * 2;
  _Float16* r2h   = (_Float16*)wp; wp += (size_t)524288 * 2;
  float*    cbn   = (float*)wp;    wp += (size_t)2048 * 4;
  int*      idxw  = (int*)wp;

  // outputs (concatenated, reference return order)
  float* out_q    = (float*)d_out;       // [16384,512]
  float* out_loss = out_q + 8388608;     // [1]
  float* out_idx  = out_loss + 1;        // [16384]
  float* out_hash = out_idx + 16384;     // [16384,64]
  float* out_reg  = out_hash + 1048576;  // [16384,2048]

  // ---- 0. weight pre-conversion to f16 (conv weights -> tap-major) ----
  cvt_f16_kernel<<<1024, 256, 0, stream>>>(W_in, winh, 262144);
  convw_kernel<<<1024, 256, 0, stream>>>(c3w, c3t, 262144, 3);
  convw_kernel<<<1024, 256, 0, stream>>>(c5w, c5t, 262144, 5);
  convw_kernel<<<1024, 256, 0, stream>>>(c7w, c7t, 262144, 7);
  convw_kernel<<<1024, 256, 0, stream>>>(c9w, c9t, 262144, 9);
  cvt_f16_kernel<<<4096, 256, 0, stream>>>(msw, mswh, 1048576);
  cvt_f16_kernel<<<3072, 256, 0, stream>>>(ipw, ipwh, 786432);
  cvt_f16_kernel<<<1024, 256, 0, stream>>>(aow, aowh, 262144);
  cvt_f16_kernel<<<4096, 256, 0, stream>>>(codebook, cbh, 1048576);
  cvt_f16_kernel<<<128, 256, 0, stream>>>(hw, hwh, 32768);
  cvt_f16_kernel<<<512, 256, 0, stream>>>(r1w, r1h, 131072);
  cvt_f16_kernel<<<2048, 256, 0, stream>>>(r2w, r2h, 524288);

  // ---- 1. x = market @ W_in^T + b_in  (f32 in -> f16 out) ----
  gemm_wmma_kernel<float, _Float16><<<dim3(8, 64), 256, 0, stream>>>(
      market, winh, b_in, nullptr, xh, M, 512, 512, 512, 0);
  // ---- 2. x += pos_enc ----
  add_pos_kernel<<<32768, 256, 0, stream>>>(xh, pos);
  // ---- 3. four tap-major convs + ReLU -> multi (f16, ld 2048) ----
  conv_wmma_kernel<<<dim3(8, 64), 256, 0, stream>>>(
      xh, c3t, c3b, multh + 0,    512, 262144, 2048, 3);
  conv_wmma_kernel<<<dim3(8, 64), 256, 0, stream>>>(
      xh, c5t, c5b, multh + 512,  512, 262144, 2048, 5);
  conv_wmma_kernel<<<dim3(8, 64), 256, 0, stream>>>(
      xh, c7t, c7b, multh + 1024, 512, 262144, 2048, 7);
  conv_wmma_kernel<<<dim3(8, 64), 256, 0, stream>>>(
      xh, c9t, c9b, multh + 1536, 512, 262144, 2048, 9);
  // ---- 4. proj = multi @ ms_proj_w^T -> projh ----
  gemm_wmma_kernel<_Float16, _Float16><<<dim3(8, 64), 256, 0, stream>>>(
      multh, mswh, msb, nullptr, projh, M, 512, 2048, 512, 0);
  // ---- 5. qkv = proj @ in_proj_w^T -> multh (multi dead) ----
  gemm_wmma_kernel<_Float16, _Float16><<<dim3(24, 64), 256, 0, stream>>>(
      projh, ipwh, ipb, nullptr, multh, M, 1536, 512, 1536, 0);
  // ---- 6. flash attention -> xh (x dead) ----
  attn_wmma_kernel<<<2048, 128, 0, stream>>>(multh, xh);
  // ---- 7. enc = proj + mha @ attn_out_w^T -> encf (qkv dead) ----
  gemm_wmma_kernel<_Float16, float><<<dim3(8, 64), 256, 0, stream>>>(
      xh, aowh, aob, projh, encf, M, 512, 512, 512, 2);
  // ---- 8. codebook norms (f32 codebook) ----
  cbnorm_kernel<<<2048, 256, 0, stream>>>(codebook, cbn, 512);
  // ---- 9. VQ argmin ----
  vq_argmin_kernel<<<128, 256, 0, stream>>>(encf, cbh, cbn, idxw, 2048, 512);
  // ---- 10-11. loss zero + gather ----
  zero_kernel<<<1, 1, 0, stream>>>(out_loss);
  vq_gather_kernel<<<16384, 256, 0, stream>>>(
      encf, codebook, idxw, out_q, out_idx, out_loss, 512,
      1.25f / (16384.f * 512.f));
  // ---- 12. hash codes = (quant @ hash_w^T > 0) ----
  gemm_wmma_kernel<float, float><<<dim3(1, 64), 256, 0, stream>>>(
      out_q, hwh, nullptr, nullptr, out_hash, M, 64, 512, 64, 3);
  // ---- 13. h = relu(quant @ reg1_w^T + b) -> xh (mha dead) ----
  gemm_wmma_kernel<float, _Float16><<<dim3(4, 64), 256, 0, stream>>>(
      out_q, r1h, r1b, nullptr, xh, M, 256, 512, 256, 1);
  // ---- 14. logits = h @ reg2_w^T + b -> regime region ----
  gemm_wmma_kernel<_Float16, float><<<dim3(32, 64), 256, 0, stream>>>(
      xh, r2h, r2b, nullptr, out_reg, M, 2048, 256, 2048, 0);
  // ---- 15. rowwise softmax in place ----
  softmax_row_kernel<<<16384, 256, 0, stream>>>(out_reg, 2048);
}
